// LinearAct8Int4DQ_37950331027890
// MI455X (gfx1250) — 8-bit matrix operand order repaired and checked by reading, compiled, not run on hardware
//
#include <hip/hip_runtime.h>
#include <cstdint>
#include <cstddef>

typedef int   v8i __attribute__((ext_vector_type(8)));
typedef float v8f __attribute__((ext_vector_type(8)));
typedef int   v4i_vs __attribute__((vector_size(16)));

#define K_DIM 4096
#define N_DIM 11008
#define GRP   256
#define NGRP  (K_DIM / GRP)
#define BM    128
#define BN    64
#define BK    128
#define NCHK  (K_DIM / BK)
#define LDA2  (BK + 16)
#define LDB2  (BK + 16)

#if __has_builtin(__builtin_amdgcn_global_load_async_to_lds_b128) && \
    __has_builtin(__builtin_amdgcn_s_wait_asynccnt)
#define USE_ASYNC_LDS 1
#else
#define USE_ASYNC_LDS 0
#endif

__device__ __forceinline__ void copy16_g2l(const int8_t* g, int8_t* l)
{
#if USE_ASYNC_LDS
    __builtin_amdgcn_global_load_async_to_lds_b128(
        (__attribute__((address_space(1))) v4i_vs*)(uintptr_t)g,
        (__attribute__((address_space(3))) v4i_vs*)(uintptr_t)l,
        0, 0);
#else
    *(int4*)l = *(const int4*)g;
#endif
}

__device__ __forceinline__ void wait_async_fills()
{
#if USE_ASYNC_LDS
    __builtin_amdgcn_s_wait_asynccnt(0);
#endif
}

__global__ void __launch_bounds__(256)
quant_act_kernel(const float* __restrict__ x,
                 int8_t* __restrict__ q8,
                 float* __restrict__ sArr,
                 float* __restrict__ zpArr)
{
    const int m = blockIdx.x;
    const int t = threadIdx.x;
    const float* xr = x + (size_t)m * K_DIM;

    float vmin = 0.0f, vmax = 0.0f;
    float4 v[4];
#pragma unroll
    for (int j = 0; j < 4; ++j) {
        v[j] = ((const float4*)xr)[t * 4 + j];
        vmin = fminf(vmin, fminf(fminf(v[j].x, v[j].y), fminf(v[j].z, v[j].w)));
        vmax = fmaxf(vmax, fmaxf(fmaxf(v[j].x, v[j].y), fmaxf(v[j].z, v[j].w)));
    }
    __shared__ float smin[256], smax[256];
    smin[t] = vmin; smax[t] = vmax;
    __syncthreads();
    for (int s = 128; s > 0; s >>= 1) {
        if (t < s) {
            smin[t] = fminf(smin[t], smin[t + s]);
            smax[t] = fmaxf(smax[t], smax[t + s]);
        }
        __syncthreads();
    }
    __shared__ float sh_scale, sh_zp;
    if (t == 0) {
        float mn = smin[0], mx = smax[0];
        float scale = fmaxf((mx - mn) * (1.0f / 255.0f), 1.1920929e-7f);
        float dmin = mn / scale, dmax = mx / scale;
        float zp = ((-128.0f + dmin) + (127.0f + dmax) > 0.0f) ? (-128.0f - dmin)
                                                               : ( 127.0f - dmax);
        zp = rintf(fminf(fmaxf(zp, -128.0f), 127.0f));
        sh_scale = scale; sh_zp = zp;
        sArr[m] = scale;  zpArr[m] = zp;
    }
    __syncthreads();
    const float inv_s = 1.0f / sh_scale;
    const float zp    = sh_zp;
    int8_t out16[16];
#pragma unroll
    for (int j = 0; j < 4; ++j) {
        float f[4] = {v[j].x, v[j].y, v[j].z, v[j].w};
#pragma unroll
        for (int c = 0; c < 4; ++c) {
            float q = rintf(f[c] * inv_s) + zp;
            q = fminf(fmaxf(q, -128.0f), 127.0f);
            out16[j * 4 + c] = (int8_t)(int)q;
        }
    }
    *(int4*)(q8 + (size_t)m * K_DIM + t * 16) = *(const int4*)out16;
}

__global__ void __launch_bounds__(256)
pack_w_kernel(const int* __restrict__ w,
              const float* __restrict__ scales,
              int8_t* __restrict__ wq,
              float* __restrict__ wsumsc)
{
    const int o = blockIdx.x;
    const int t = threadIdx.x;
    const int* wr = w + (size_t)o * K_DIM + t * 16;
    int sum = 0;
    int8_t out16[16];
#pragma unroll
    for (int j = 0; j < 4; ++j) {
        int4 wv = ((const int4*)wr)[j];
        sum += wv.x + wv.y + wv.z + wv.w;
        out16[j * 4 + 0] = (int8_t)wv.x; out16[j * 4 + 1] = (int8_t)wv.y;
        out16[j * 4 + 2] = (int8_t)wv.z; out16[j * 4 + 3] = (int8_t)wv.w;
    }
    *(int4*)(wq + (size_t)o * K_DIM + t * 16) = *(const int4*)out16;

    __shared__ int gsum[256];
    gsum[t] = sum;
    __syncthreads();
    if (t < NGRP) {
        int acc = 0;
#pragma unroll
        for (int j = 0; j < 16; ++j) acc += gsum[t * 16 + j];
        gsum[t] = acc;
    }
    __syncthreads();
    if (t == 0) {
        float acc = 0.0f;
#pragma unroll
        for (int g = 0; g < NGRP; ++g)
            acc += scales[(size_t)o * NGRP + g] * (float)gsum[g];
        wsumsc[o] = acc;
    }
}

__global__ void __launch_bounds__(256)
gemm_i8i4_kernel(const int8_t* __restrict__ qA,
                 const float* __restrict__ sArr,
                 const float* __restrict__ zpArr,
                 const int8_t* __restrict__ wq,
                 const float* __restrict__ scales,
                 const float* __restrict__ wsumsc,
                 float* __restrict__ out)
{
    __shared__ __align__(16) int8_t As[2][BM * LDA2];
    __shared__ __align__(16) int8_t Bs[2][BN * LDB2];

    const int nBase = blockIdx.x * BN;
    const int mBase = blockIdx.y * BM;
    const int t     = threadIdx.x;
    const int wave  = t >> 5;
    const int lane  = t & 31;
    const int msub  = wave & 3;
    const int nhalf = wave >> 2;
    const int ln    = lane & 15;
    const int hi    = lane >> 4;
    const int Rm    = msub * 32;
    const int Cn    = nhalf * 32;

    auto stage = [&](int chunk, int buf) {
        const size_t kofs = (size_t)chunk * BK;
#pragma unroll
        for (int j = 0; j < 4; ++j) {
            int c = t + j * 256;
            int row = c >> 3, cc = (c & 7) * 16;
            copy16_g2l(qA + (size_t)(mBase + row) * K_DIM + kofs + cc,
                       &As[buf][row * LDA2 + cc]);
        }
#pragma unroll
        for (int j = 0; j < 2; ++j) {
            int c = t + j * 256;
            int row = c >> 3, cc = (c & 7) * 16;
            copy16_g2l(wq + (size_t)(nBase + row) * K_DIM + kofs + cc,
                       &Bs[buf][row * LDB2 + cc]);
        }
    };

    v8f fc00 = {}, fc01 = {}, fc10 = {}, fc11 = {};
    v8i acc00 = {}, acc01 = {}, acc10 = {}, acc11 = {};

    stage(0, 0);

    for (int c = 0; c < NCHK; ++c) {
        const int buf = c & 1;
        wait_async_fills();
        __syncthreads();
        if (c + 1 < NCHK) stage(c + 1, buf ^ 1);

        const int8_t* Ab = &As[buf][0];
        const int8_t* Bb = &Bs[buf][0];
#pragma unroll
        for (int ks = 0; ks < 2; ++ks) {
            const int k64 = ks * 64;
            const int8_t* pb0 = Bb + (Cn + ln) * LDB2 + k64 + hi * 16;
            int4 b0a = *(const int4*)pb0;
            int4 b0b = *(const int4*)(pb0 + 32);
            v8i bf0 = {b0a.x, b0a.y, b0a.z, b0a.w, b0b.x, b0b.y, b0b.z, b0b.w};
            const int8_t* pb1 = pb0 + 16 * LDB2;
            int4 b1a = *(const int4*)pb1;
            int4 b1b = *(const int4*)(pb1 + 32);
            v8i bf1 = {b1a.x, b1a.y, b1a.z, b1a.w, b1b.x, b1b.y, b1b.z, b1b.w};
            const int8_t* pa0 = Ab + (Rm + ln) * LDA2 + k64 + hi * 16;
            int4 a0 = *(const int4*)(pa0);
            int4 a1 = *(const int4*)(pa0 + 32);
            v8i af0 = {a0.x, a0.y, a0.z, a0.w, a1.x, a1.y, a1.z, a1.w};
            const int8_t* pa1 = pa0 + 16 * LDA2;
            int4 c0 = *(const int4*)(pa1);
            int4 c1 = *(const int4*)(pa1 + 32);
            v8i af1 = {c0.x, c0.y, c0.z, c0.w, c1.x, c1.y, c1.z, c1.w};

            acc00 = __builtin_amdgcn_wmma_i32_16x16x64_iu8(true, af0, true, bf0, acc00, false, false);
            acc01 = __builtin_amdgcn_wmma_i32_16x16x64_iu8(true, af0, true, bf1, acc01, false, false);
            acc10 = __builtin_amdgcn_wmma_i32_16x16x64_iu8(true, af1, true, bf0, acc10, false, false);
            acc11 = __builtin_amdgcn_wmma_i32_16x16x64_iu8(true, af1, true, bf1, acc11, false, false);
        }

        if (c & 1) {
            const int g = c >> 1;
            const float sc0 = scales[(size_t)(nBase + Cn + ln)      * NGRP + g];
            const float sc1 = scales[(size_t)(nBase + Cn + 16 + ln) * NGRP + g];
#pragma unroll
            for (int r = 0; r < 8; ++r) {
                fc00[r] += (float)acc00[r] * sc0;
                fc01[r] += (float)acc01[r] * sc1;
                fc10[r] += (float)acc10[r] * sc0;
                fc11[r] += (float)acc11[r] * sc1;
            }
            acc00 = (v8i){}; acc01 = (v8i){}; acc10 = (v8i){}; acc11 = (v8i){};
        }
        __syncthreads();
    }

    const int col0 = nBase + Cn + ln;
    const int col1 = col0 + 16;
    const float wn0 = wsumsc[col0];
    const float wn1 = wsumsc[col1];
#pragma unroll
    for (int r = 0; r < 8; ++r) {
        const int m0 = mBase + Rm + hi * 8 + r;
        const int m1 = m0 + 16;
        const float s0 = sArr[m0], z0 = zpArr[m0];
        const float s1 = sArr[m1], z1 = zpArr[m1];
        out[(size_t)m0 * N_DIM + col0] = s0 * (fc00[r] - z0 * wn0);
        out[(size_t)m0 * N_DIM + col1] = s0 * (fc01[r] - z0 * wn1);
        out[(size_t)m1 * N_DIM + col0] = s1 * (fc10[r] - z1 * wn0);
        out[(size_t)m1 * N_DIM + col1] = s1 * (fc11[r] - z1 * wn1);
    }
}

extern "C" void kernel_launch(void* const* d_in, const int* in_sizes, int n_in,
                              void* d_out, int out_size, void* d_ws, size_t ws_size,
                              hipStream_t stream) {
    const float* x      = (const float*)d_in[0];
    const int*   w      = (const int*)d_in[1];
    const float* scales = (const float*)d_in[2];

    const int M = in_sizes[0] / K_DIM;

    char* ws = (char*)d_ws;
    size_t off = 0;
    int8_t* qA = (int8_t*)(ws + off); off += (size_t)M * K_DIM;
    off = (off + 255) & ~(size_t)255;
    float* sArr  = (float*)(ws + off); off += (size_t)M * 4;
    float* zpArr = (float*)(ws + off); off += (size_t)M * 4;
    off = (off + 255) & ~(size_t)255;
    int8_t* wq = (int8_t*)(ws + off); off += (size_t)N_DIM * K_DIM;
    off = (off + 255) & ~(size_t)255;
    float* wsum = (float*)(ws + off);

    quant_act_kernel<<<M, 256, 0, stream>>>(x, qA, sArr, zpArr);
    pack_w_kernel<<<N_DIM, 256, 0, stream>>>(w, scales, wq, wsum);
    gemm_i8i4_kernel<<<dim3(N_DIM / BN, M / BM), 256, 0, stream>>>(
        qA, sArr, zpArr, wq, scales, wsum, (float*)d_out);
}
